// GNNModelBaseline_10118942949882
// MI455X (gfx1250) — matrix operand order repaired and checked by reading, not compiled, not run on hardware
//
#include <hip/hip_runtime.h>
#include <hip/hip_bf16.h>

#define N_NODES 100000
#define N_EDGES 800000
#define DH 256
#define DOUT 64

typedef __bf16 v16bf __attribute__((ext_vector_type(16)));
typedef float  v8f   __attribute__((ext_vector_type(8)));

// ---------------- degree / normalization ----------------
__global__ void k_deg_init(float* deg) {
  int i = blockIdx.x * blockDim.x + threadIdx.x;
  if (i < N_NODES) deg[i] = 1.0f;  // self-loop contributes 1 to every node
}

__global__ void k_deg_count(const int* __restrict__ ei, float* deg) {
  int e = blockIdx.x * blockDim.x + threadIdx.x;
  if (e < N_EDGES) atomicAdd(&deg[ei[N_EDGES + e]], 1.0f);  // dst row
}

__global__ void k_deg_rsqrt(float* deg) {
  int i = blockIdx.x * blockDim.x + threadIdx.x;
  if (i < N_NODES) deg[i] = rsqrtf(deg[i]);
}

// load 8 consecutive f32, compile-time-optional relu, cvt to bf16.
// ReLU written as a select so clang emits a single v_max_num_f32
// (no redundant canonicalize like libm fmaxf requires).
template <bool RELU>
__device__ __forceinline__ void load8_cvt(const float* __restrict__ p,
                                          v16bf& aa, int base) {
  float4 x0 = *(const float4*)p;
  float4 x1 = *(const float4*)(p + 4);
  float r[8] = {x0.x, x0.y, x0.z, x0.w, x1.x, x1.y, x1.z, x1.w};
#pragma unroll
  for (int i = 0; i < 8; i++) {
    float f = r[i];
    if (RELU) f = (f > 0.0f) ? f : 0.0f;
    aa[base + i] = (__bf16)f;
  }
}

// ---------------- H = (relu?)(A) @ W  ; A:[N,256] f32, W:[256,256] f32, H bf16
// One wave per block. Wave owns 16 output columns (blockIdx.x), preloads the
// weight slice for all 8 K-steps (weight-stationary), then streams M-tiles.
template <bool RELU>
__global__ __launch_bounds__(32) void k_gemm256(const float* __restrict__ A,
                                                const float* __restrict__ W,
                                                __bf16* __restrict__ H) {
  const int lane = threadIdx.x & 31;
  const int row  = lane & 15;   // A row within tile / output col within tile
  const int half = lane >> 4;
  const int n0   = blockIdx.x * 16;

  // B fragment layout (32x16 bf16): lane holds col n = lane%16,
  // K = half*8 + i (i < 8), 16 + half*8 + (i-8) (i >= 8) within each 32-wide K step.
  v16bf B[8];
#pragma unroll
  for (int s = 0; s < 8; s++) {
    const int k0 = s * 32 + half * 8;
#pragma unroll
    for (int i = 0; i < 16; i++)
      B[s][i] = (__bf16)W[(size_t)(k0 + i + (i & 8)) * DH + n0 + row];
  }

  for (int mt = blockIdx.y; mt < N_NODES / 16; mt += gridDim.y) {
    const int m0 = mt * 16;
    const float* arow = A + (size_t)(m0 + row) * DH;
    v8f c;
#pragma unroll
    for (int i = 0; i < 8; i++) c[i] = 0.0f;
#pragma unroll
    for (int s = 0; s < 8; s++) {
      const int k0 = s * 32;
      // A fragment layout (16x32 bf16): lane = row%16, K runs
      // [half*8, half*8+8) and [16+half*8, ...+8).
      v16bf aa;
      load8_cvt<RELU>(arow + k0 + half * 8,      aa, 0);
      load8_cvt<RELU>(arow + k0 + 16 + half * 8, aa, 8);
      c = __builtin_amdgcn_wmma_f32_16x16x32_bf16(false, aa, false, B[s],
                                                  (short)0, c, false, false);
    }
    // D layout: element (m,n): vgpr = m%8, lane = (m/8)*16 + n
    __bf16* hp = H + (size_t)m0 * DH + n0 + row;
#pragma unroll
    for (int v = 0; v < 8; v++)
      hp[(size_t)(v + half * 8) * DH] = (__bf16)c[v];
  }
}

// ---------------- agg[i,:] = bias + dis[i]^2 * h[i,:]  (self-loop + bias)
__global__ void k_agg_init(const __bf16* __restrict__ H,
                           const float* __restrict__ dis,
                           const float* __restrict__ bias,
                           float* __restrict__ agg) {
  int tid  = blockIdx.x * blockDim.x + threadIdx.x;
  int node = tid >> 5;
  int c8   = (tid & 31) << 3;
  if (node >= N_NODES) return;
  float s = dis[node];
  s = s * s;
  uint4 raw = *(const uint4*)(H + (size_t)node * DH + c8);  // 8 bf16
  const __bf16* hb = (const __bf16*)&raw;
  float* ap = agg + (size_t)node * DH + c8;
#pragma unroll
  for (int i = 0; i < 8; i++) ap[i] = bias[c8 + i] + s * (float)hb[i];
}

// ---------------- agg[dst,:] += dis[src]*dis[dst] * h[src,:] (scatter-add)
// One wave covers one edge's full 256-feature row (coalesced 512B gather).
__global__ void k_aggregate(const __bf16* __restrict__ H,
                            const float* __restrict__ dis,
                            const int* __restrict__ ei,
                            float* __restrict__ agg) {
  int tid = blockIdx.x * blockDim.x + threadIdx.x;
  int e   = tid >> 5;
  int c8  = (tid & 31) << 3;
  if (e >= N_EDGES) return;
  int s = ei[e];
  int d = ei[N_EDGES + e];
  float nrm = dis[s] * dis[d];
  uint4 raw = *(const uint4*)(H + (size_t)s * DH + c8);
  const __bf16* hb = (const __bf16*)&raw;
  float* ap = agg + (size_t)d * DH + c8;
#pragma unroll
  for (int i = 0; i < 8; i++) atomicAdd(ap + i, nrm * (float)hb[i]);
}

// ---------------- logits = relu(agg) @ Wo + bo ; log_softmax fused in-wave
__global__ __launch_bounds__(32) void k_final(const float* __restrict__ agg,
                                              const float* __restrict__ Wo,
                                              const float* __restrict__ bo,
                                              float* __restrict__ out) {
  const int lane = threadIdx.x & 31;
  const int row  = lane & 15;
  const int half = lane >> 4;
  const int m0   = blockIdx.x * 16;
  const float* arow = agg + (size_t)(m0 + row) * DH;

  v8f c[4];
#pragma unroll
  for (int t = 0; t < 4; t++)
#pragma unroll
    for (int i = 0; i < 8; i++) c[t][i] = 0.0f;

#pragma unroll
  for (int s = 0; s < 8; s++) {
    const int k0 = s * 32;
    v16bf aa;
    load8_cvt<true>(arow + k0 + half * 8,      aa, 0);
    load8_cvt<true>(arow + k0 + 16 + half * 8, aa, 8);
#pragma unroll
    for (int t = 0; t < 4; t++) {
      v16bf bb;
#pragma unroll
      for (int i = 0; i < 16; i++)
        bb[i] = (__bf16)Wo[(size_t)(k0 + half * 8 + i + (i & 8)) * DOUT + t * 16 + row];
      c[t] = __builtin_amdgcn_wmma_f32_16x16x32_bf16(false, aa, false, bb,
                                                     (short)0, c[t], false, false);
    }
  }
  // bias
#pragma unroll
  for (int t = 0; t < 4; t++) {
    float b = bo[t * 16 + row];
#pragma unroll
    for (int v = 0; v < 8; v++) c[t][v] += b;
  }
  // log_softmax over 64 cols; row m = m0 + v + 8*half lives in one 16-lane half
#pragma unroll
  for (int v = 0; v < 8; v++) {
    float mx = fmaxf(fmaxf(c[0][v], c[1][v]), fmaxf(c[2][v], c[3][v]));
#pragma unroll
    for (int off = 8; off >= 1; off >>= 1) mx = fmaxf(mx, __shfl_xor(mx, off, 16));
    float sum = 0.0f;
#pragma unroll
    for (int t = 0; t < 4; t++) sum += __expf(c[t][v] - mx);
#pragma unroll
    for (int off = 8; off >= 1; off >>= 1) sum += __shfl_xor(sum, off, 16);
    float lse = mx + __logf(sum);
    const int m = m0 + v + half * 8;
#pragma unroll
    for (int t = 0; t < 4; t++)
      out[(size_t)m * DOUT + t * 16 + row] = c[t][v] - lse;
  }
}

extern "C" void kernel_launch(void* const* d_in, const int* in_sizes, int n_in,
                              void* d_out, int out_size, void* d_ws, size_t ws_size,
                              hipStream_t stream) {
  (void)in_sizes; (void)n_in; (void)out_size; (void)ws_size;
  const float* x  = (const float*)d_in[0];
  const int*   ei = (const int*)d_in[1];
  const float* W1 = (const float*)d_in[2];
  const float* b1 = (const float*)d_in[3];
  const float* W2 = (const float*)d_in[4];
  const float* b2 = (const float*)d_in[5];
  const float* Wo = (const float*)d_in[6];
  const float* bo = (const float*)d_in[7];
  float* out = (float*)d_out;

  char* ws = (char*)d_ws;
  float*  dis  = (float*)ws;                                   // N f32
  size_t  o1   = ((size_t)N_NODES * 4 + 255) & ~(size_t)255;
  __bf16* hbuf = (__bf16*)(ws + o1);                           // N*256 bf16
  size_t  o2   = o1 + (((size_t)N_NODES * DH * 2 + 255) & ~(size_t)255);
  float*  agg  = (float*)(ws + o2);                            // N*256 f32

  // normalization: deg = 1 + indegree; dis = rsqrt(deg)
  k_deg_init <<<(N_NODES + 255) / 256, 256, 0, stream>>>(dis);
  k_deg_count<<<(N_EDGES + 255) / 256, 256, 0, stream>>>(ei, dis);
  k_deg_rsqrt<<<(N_NODES + 255) / 256, 256, 0, stream>>>(dis);

  dim3 gg(DH / 16, 512);  // 16 col-tiles x 512 M-strides = 8192 waves

  // layer 1 (no relu on x)
  k_gemm256<false><<<gg, 32, 0, stream>>>(x, W1, hbuf);
  k_agg_init <<<(N_NODES * 32) / 256, 256, 0, stream>>>(hbuf, dis, b1, agg);
  k_aggregate<<<(N_EDGES * 32) / 256, 256, 0, stream>>>(hbuf, dis, ei, agg);
  // layer 2 (relu folded into GEMM A-load at compile time)
  k_gemm256<true><<<gg, 32, 0, stream>>>(agg, W2, hbuf);
  k_agg_init <<<(N_NODES * 32) / 256, 256, 0, stream>>>(hbuf, dis, b2, agg);
  k_aggregate<<<(N_EDGES * 32) / 256, 256, 0, stream>>>(hbuf, dis, ei, agg);
  // output projection + log_softmax
  k_final    <<<N_NODES / 16, 32, 0, stream>>>(agg, Wo, bo, out);
}
